// SpatialGNN_40836549050600
// MI455X (gfx1250) — matrix operand order repaired and checked by reading, not compiled, not run on hardware
//
#include <hip/hip_runtime.h>
#include <hip/hip_bf16.h>
#include <math.h>

typedef __bf16 bf16_t;
typedef __attribute__((ext_vector_type(16))) __bf16 v16bf;
typedef __attribute__((ext_vector_type(8)))  __bf16 v8bf;
typedef __attribute__((ext_vector_type(8)))  float  v8f;

#define NEG_SLOPE 0.2f

// ---------------------------------------------------------------------------
// helpers
// ---------------------------------------------------------------------------
__device__ __forceinline__ void atomicMaxF(float* addr, float val) {
  // monotone mapping trick: works for all finite floats and -inf init
  if (val >= 0.0f) atomicMax((int*)addr, __float_as_int(val));
  else             atomicMin((unsigned int*)addr, __float_as_uint(val));
}

__device__ __forceinline__ void edgePair(const long long* __restrict__ ei,
                                         int E, int e, int& s, int& d) {
  if (e < E) { s = (int)ei[e]; d = (int)ei[(size_t)E + e]; }
  else       { s = d = e - E; }                         // self loops
}

// ---------------------------------------------------------------------------
// fills / converts
// ---------------------------------------------------------------------------
__global__ void fill_f32(float* __restrict__ p, float v, int n) {
  int i = blockIdx.x * blockDim.x + threadIdx.x;
  if (i < n) p[i] = v;
}

__global__ void cvt_f32_bf16(const float* __restrict__ in, bf16_t* __restrict__ out, int n) {
  int i = blockIdx.x * blockDim.x + threadIdx.x;
  if (i < n) out[i] = (bf16_t)in[i];
}

// pack W[K, Nout] (row major) into WMMA B-operand lane layout:
// Bp[((ct*KB + kb)*32 + lane)*16 + j] = W[kb*32 + (lane>=16)*16 + j][ct*16 + (lane&15)]
__global__ void pack_w_bf16(const float* __restrict__ W, bf16_t* __restrict__ Bp,
                            int K, int Nout, int total) {
  int t = blockIdx.x * blockDim.x + threadIdx.x;
  if (t >= total) return;
  const int KB   = K >> 5;
  const int j    = t & 15;
  const int rest = t >> 4;
  const int lane = rest & 31;
  const int r2   = rest >> 5;
  const int kb   = r2 % KB;
  const int ct   = r2 / KB;
  const int k = kb * 32 + (lane >> 4) * 16 + j;
  const int n = (ct << 4) + (lane & 15);
  Bp[t] = (bf16_t)W[(size_t)k * Nout + n];
}

// ---------------------------------------------------------------------------
// bf16 WMMA GEMM: C[M,Nout] = A[M,K] * W (W pre-packed). One wave = 16x16 tile.
// ---------------------------------------------------------------------------
__global__ void gemm_bf16_wmma(const bf16_t* __restrict__ A,
                               const bf16_t* __restrict__ Bp,
                               float* __restrict__ C,
                               int M, int K, int Nout) {
  const int lane    = threadIdx.x & 31;
  const int wave    = threadIdx.x >> 5;
  const int colTile = blockIdx.x * 8 + wave;
  const int nTiles  = Nout >> 4;
  if (colTile >= nTiles) return;
  const int row0 = blockIdx.y << 4;
  const int KB   = K >> 5;
  const int half = lane >> 4;

  int r = row0 + (lane & 15);
  if (r >= M) r = M - 1;                     // clamp; stores are guarded below
  const bf16_t* __restrict__ arow  = A  + (size_t)r * K + half * 16;
  const bf16_t* __restrict__ bbase = Bp + ((size_t)colTile * KB) * 512 + (size_t)lane * 16;

  v8f acc = {};
  for (int kb = 0; kb < KB; ++kb) {
    union { v16bf v; v8bf h[2]; } a;
    a.h[0] = *(const v8bf*)(arow + kb * 32);        // K[half*16 .. +7]
    a.h[1] = *(const v8bf*)(arow + kb * 32 + 8);    // K[half*16+8 .. +7]
    v16bf b = *(const v16bf*)(bbase + (size_t)kb * 512);
    acc = __builtin_amdgcn_wmma_f32_16x16x32_bf16(false, a.v, false, b,
                                                  (short)0, acc, false, false);
  }

  const int col   = (colTile << 4) + (lane & 15);
  const int rbase = row0 + (half ? 8 : 0);
#pragma unroll
  for (int v = 0; v < 8; ++v) {
    int rr = rbase + v;
    if (rr < M) C[(size_t)rr * Nout + col] = acc[v];
  }
}

// ---------------------------------------------------------------------------
// attention coefficients: a_src[n,h] = <h[n,h,:], att_src[h,:]> (C=128, H=4)
// one wave per (n,h); lane covers 4 channels via float4
// ---------------------------------------------------------------------------
__global__ void att_reduce(const float* __restrict__ h,
                           const float* __restrict__ att_src,
                           const float* __restrict__ att_dst,
                           float* __restrict__ a_src, float* __restrict__ a_dst,
                           int N) {
  const int gw   = (blockIdx.x * blockDim.x + threadIdx.x) >> 5;
  const int lane = threadIdx.x & 31;
  if (gw >= N * 4) return;
  const int n = gw >> 2, hh = gw & 3;
  const int base = hh * 128 + lane * 4;
  const float4 v  = *(const float4*)(h + (size_t)n * 512 + base);
  const float4 ws = *(const float4*)(att_src + base);
  const float4 wd = *(const float4*)(att_dst + base);
  float ds = v.x * ws.x + v.y * ws.y + v.z * ws.z + v.w * ws.w;
  float dd = v.x * wd.x + v.y * wd.y + v.z * wd.z + v.w * wd.w;
#pragma unroll
  for (int o = 16; o > 0; o >>= 1) {
    ds += __shfl_xor(ds, o, 32);
    dd += __shfl_xor(dd, o, 32);
  }
  if (lane == 0) { a_src[gw] = ds; a_dst[gw] = dd; }
}

// ---------------------------------------------------------------------------
// edge logits (leaky relu) + atomic segment max  (thread per (edge, head))
// ---------------------------------------------------------------------------
__global__ void edge_logits_max(const float* __restrict__ a_src,
                                const float* __restrict__ a_dst,
                                const long long* __restrict__ ei, int E, int Etot,
                                float* __restrict__ logit, float* __restrict__ m) {
  int t = blockIdx.x * blockDim.x + threadIdx.x;
  if (t >= Etot * 4) return;
  int e = t >> 2, hh = t & 3, s, d;
  edgePair(ei, E, e, s, d);
  float l = a_src[s * 4 + hh] + a_dst[d * 4 + hh];
  l = l > 0.0f ? l : NEG_SLOPE * l;
  logit[t] = l;
  atomicMaxF(m + d * 4 + hh, l);
}

__global__ void edge_exp_sum(const float* __restrict__ m,
                             const long long* __restrict__ ei, int E, int Etot,
                             float* __restrict__ elog, float* __restrict__ ssum) {
  int t = blockIdx.x * blockDim.x + threadIdx.x;
  if (t >= Etot * 4) return;
  int e = t >> 2, hh = t & 3, s, d;
  edgePair(ei, E, e, s, d);
  float v = expf(elog[t] - m[d * 4 + hh]);
  elog[t] = v;
  atomicAdd(ssum + d * 4 + hh, v);
}

// ---------------------------------------------------------------------------
// weighted scatter: agg[dst] += h[src] * alpha   (wave per edge, 512 channels)
// ---------------------------------------------------------------------------
__global__ void aggregate(const float* __restrict__ hpre,
                          const float* __restrict__ elog,
                          const float* __restrict__ ssum,
                          const long long* __restrict__ ei, int E, int Etot,
                          float* __restrict__ agg) {
  const int gw   = (blockIdx.x * blockDim.x + threadIdx.x) >> 5;
  const int lane = threadIdx.x & 31;
  if (gw >= Etot) return;
  int s, d;
  edgePair(ei, E, gw, s, d);
#pragma unroll
  for (int hh = 0; hh < 4; ++hh) {
    const float alpha = elog[(size_t)gw * 4 + hh] / (ssum[(size_t)d * 4 + hh] + 1e-16f);
    const int col = hh * 128 + lane * 4;
    const float4 v = *(const float4*)(hpre + (size_t)s * 512 + col);
    float* o = agg + (size_t)d * 512 + col;
    atomicAdd(o + 0, v.x * alpha);
    atomicAdd(o + 1, v.y * alpha);
    atomicAdd(o + 2, v.z * alpha);
    atomicAdd(o + 3, v.w * alpha);
  }
}

// bias + relu + convert to bf16 for the next GEMM
__global__ void finalize_relu_bf16(const float* __restrict__ agg,
                                   const float* __restrict__ bias,
                                   bf16_t* __restrict__ out, int total) {
  int t = blockIdx.x * blockDim.x + threadIdx.x;
  if (t >= total) return;
  float v = agg[t] + bias[t & 511];
  out[t] = (bf16_t)(v > 0.0f ? v : 0.0f);
}

// ---------------------------------------------------------------------------
// layer 3: fused relu(agg2 + b2) @ W3 -> h3[N,10]
// ---------------------------------------------------------------------------
__global__ void gemm3_fused(const float* __restrict__ agg2,
                            const float* __restrict__ b2,
                            const float* __restrict__ w3,
                            float* __restrict__ h3, int N) {
  int t = blockIdx.x * blockDim.x + threadIdx.x;
  if (t >= N * 10) return;
  const int n = t / 10, c = t % 10;
  const float* __restrict__ row = agg2 + (size_t)n * 512;
  float acc = 0.0f;
  for (int k = 0; k < 512; ++k) {
    float v = row[k] + b2[k];
    v = v > 0.0f ? v : 0.0f;
    acc += v * w3[k * 10 + c];
  }
  h3[t] = acc;
}

__global__ void att3_reduce(const float* __restrict__ h3,
                            const float* __restrict__ att_src,
                            const float* __restrict__ att_dst,
                            float* __restrict__ a3s, float* __restrict__ a3d, int N) {
  int n = blockIdx.x * blockDim.x + threadIdx.x;
  if (n >= N) return;
  float ds = 0.0f, dd = 0.0f;
#pragma unroll
  for (int c = 0; c < 10; ++c) {
    float v = h3[n * 10 + c];
    ds += v * att_src[c];
    dd += v * att_dst[c];
  }
  a3s[n] = ds; a3d[n] = dd;
}

__global__ void l3_logits_max(const float* __restrict__ a3s, const float* __restrict__ a3d,
                              const long long* __restrict__ ei, int E, int Etot,
                              float* __restrict__ e3, float* __restrict__ m3) {
  int t = blockIdx.x * blockDim.x + threadIdx.x;
  if (t >= Etot) return;
  int s, d; edgePair(ei, E, t, s, d);
  float l = a3s[s] + a3d[d];
  l = l > 0.0f ? l : NEG_SLOPE * l;
  e3[t] = l;
  atomicMaxF(m3 + d, l);
}

__global__ void l3_exp_sum(const float* __restrict__ m3,
                           const long long* __restrict__ ei, int E, int Etot,
                           float* __restrict__ e3, float* __restrict__ s3) {
  int t = blockIdx.x * blockDim.x + threadIdx.x;
  if (t >= Etot) return;
  int s, d; edgePair(ei, E, t, s, d);
  float v = expf(e3[t] - m3[d]);
  e3[t] = v;
  atomicAdd(s3 + d, v);
}

__global__ void l3_aggregate(const float* __restrict__ h3,
                             const float* __restrict__ e3, const float* __restrict__ s3,
                             const long long* __restrict__ ei, int E, int Etot,
                             float* __restrict__ agg3) {
  int t = blockIdx.x * blockDim.x + threadIdx.x;
  if (t >= Etot) return;
  int s, d; edgePair(ei, E, t, s, d);
  float alpha = e3[t] / (s3[d] + 1e-16f);
#pragma unroll
  for (int c = 0; c < 10; ++c)
    atomicAdd(agg3 + (size_t)d * 10 + c, h3[(size_t)s * 10 + c] * alpha);
}

__global__ void final_log_softmax(const float* __restrict__ agg3,
                                  const float* __restrict__ b3,
                                  float* __restrict__ out, int N) {
  int n = blockIdx.x * blockDim.x + threadIdx.x;
  if (n >= N) return;
  float z[10], mx = -INFINITY;
#pragma unroll
  for (int c = 0; c < 10; ++c) { z[c] = agg3[n * 10 + c] + b3[c]; mx = fmaxf(mx, z[c]); }
  float sum = 0.0f;
#pragma unroll
  for (int c = 0; c < 10; ++c) sum += expf(z[c] - mx);
  float lse = logf(sum);
#pragma unroll
  for (int c = 0; c < 10; ++c) out[n * 10 + c] = z[c] - mx - lse;
}

// ---------------------------------------------------------------------------
// host side
// ---------------------------------------------------------------------------
static inline unsigned blk(long long n, int b) { return (unsigned)((n + b - 1) / b); }

extern "C" void kernel_launch(void* const* d_in, const int* in_sizes, int n_in,
                              void* d_out, int out_size, void* d_ws, size_t ws_size,
                              hipStream_t stream) {
  const float*     x   = (const float*)d_in[0];
  const long long* ei  = (const long long*)d_in[1];
  const float*     w1  = (const float*)d_in[2];
  const float*     as1 = (const float*)d_in[3];
  const float*     ad1 = (const float*)d_in[4];
  const float*     b1  = (const float*)d_in[5];
  const float*     w2  = (const float*)d_in[6];
  const float*     as2 = (const float*)d_in[7];
  const float*     ad2 = (const float*)d_in[8];
  const float*     b2  = (const float*)d_in[9];
  const float*     w3  = (const float*)d_in[10];
  const float*     as3 = (const float*)d_in[11];
  const float*     ad3 = (const float*)d_in[12];
  const float*     b3  = (const float*)d_in[13];

  const int N    = in_sizes[0] / 256;   // 20000
  const int E    = in_sizes[1] / 2;     // 320000
  const int Etot = E + N;               // self loops appended

  // ---- workspace carve-up (256B aligned regions) ----
  size_t off = 0;
  char* ws = (char*)d_ws;
  auto carve = [&](size_t bytes) -> char* {
    char* p = ws + off;
    off = (off + bytes + 255) & ~(size_t)255;
    return p;
  };
  bf16_t* w1p  = (bf16_t*)carve((size_t)256 * 512 * 2);
  bf16_t* w2p  = (bf16_t*)carve((size_t)512 * 512 * 2);
  bf16_t* xbf  = (bf16_t*)carve((size_t)N * 256 * 2);
  bf16_t* x2bf = (bf16_t*)carve((size_t)N * 512 * 2);
  float*  hpre = (float*)carve((size_t)N * 512 * 4);
  float*  agg  = (float*)carve((size_t)N * 512 * 4);
  float*  asb  = (float*)carve((size_t)N * 4 * 4);
  float*  adb  = (float*)carve((size_t)N * 4 * 4);
  float*  mb   = (float*)carve((size_t)N * 4 * 4);
  float*  sb   = (float*)carve((size_t)N * 4 * 4);
  float*  elog = (float*)carve((size_t)Etot * 4 * 4);
  float*  h3   = (float*)carve((size_t)N * 10 * 4);
  float*  a3s  = (float*)carve((size_t)N * 4);
  float*  a3d  = (float*)carve((size_t)N * 4);
  float*  m3   = (float*)carve((size_t)N * 4);
  float*  s3   = (float*)carve((size_t)N * 4);
  float*  e3   = (float*)carve((size_t)Etot * 4);
  float*  agg3 = (float*)carve((size_t)N * 10 * 4);

  const int B = 256;

  // ---- weight packing + input conversion ----
  pack_w_bf16<<<blk(256 * 512, B), B, 0, stream>>>(w1, w1p, 256, 512, 256 * 512);
  pack_w_bf16<<<blk(512 * 512, B), B, 0, stream>>>(w2, w2p, 512, 512, 512 * 512);
  cvt_f32_bf16<<<blk((long long)N * 256, B), B, 0, stream>>>(x, xbf, N * 256);

  dim3 gemmGrid(4 /*512/16/8*/, blk(N, 16));

  // =================== layer 1 ===================
  gemm_bf16_wmma<<<gemmGrid, B, 0, stream>>>(xbf, w1p, hpre, N, 256, 512);
  att_reduce<<<blk((long long)N * 4 * 32, B), B, 0, stream>>>(hpre, as1, ad1, asb, adb, N);
  fill_f32<<<blk(N * 4, B), B, 0, stream>>>(mb, -INFINITY, N * 4);
  fill_f32<<<blk(N * 4, B), B, 0, stream>>>(sb, 0.0f, N * 4);
  fill_f32<<<blk((long long)N * 512, B), B, 0, stream>>>(agg, 0.0f, N * 512);
  edge_logits_max<<<blk((long long)Etot * 4, B), B, 0, stream>>>(asb, adb, ei, E, Etot, elog, mb);
  edge_exp_sum<<<blk((long long)Etot * 4, B), B, 0, stream>>>(mb, ei, E, Etot, elog, sb);
  aggregate<<<blk((long long)Etot * 32, B), B, 0, stream>>>(hpre, elog, sb, ei, E, Etot, agg);
  finalize_relu_bf16<<<blk((long long)N * 512, B), B, 0, stream>>>(agg, b1, x2bf, N * 512);

  // =================== layer 2 ===================
  gemm_bf16_wmma<<<gemmGrid, B, 0, stream>>>(x2bf, w2p, hpre, N, 512, 512);
  att_reduce<<<blk((long long)N * 4 * 32, B), B, 0, stream>>>(hpre, as2, ad2, asb, adb, N);
  fill_f32<<<blk(N * 4, B), B, 0, stream>>>(mb, -INFINITY, N * 4);
  fill_f32<<<blk(N * 4, B), B, 0, stream>>>(sb, 0.0f, N * 4);
  fill_f32<<<blk((long long)N * 512, B), B, 0, stream>>>(agg, 0.0f, N * 512);
  edge_logits_max<<<blk((long long)Etot * 4, B), B, 0, stream>>>(asb, adb, ei, E, Etot, elog, mb);
  edge_exp_sum<<<blk((long long)Etot * 4, B), B, 0, stream>>>(mb, ei, E, Etot, elog, sb);
  aggregate<<<blk((long long)Etot * 32, B), B, 0, stream>>>(hpre, elog, sb, ei, E, Etot, agg);

  // =================== layer 3 ===================
  gemm3_fused<<<blk((long long)N * 10, B), B, 0, stream>>>(agg, b2, w3, h3, N);
  att3_reduce<<<blk(N, B), B, 0, stream>>>(h3, as3, ad3, a3s, a3d, N);
  fill_f32<<<blk(N, B), B, 0, stream>>>(m3, -INFINITY, N);
  fill_f32<<<blk(N, B), B, 0, stream>>>(s3, 0.0f, N);
  fill_f32<<<blk((long long)N * 10, B), B, 0, stream>>>(agg3, 0.0f, N * 10);
  l3_logits_max<<<blk(Etot, B), B, 0, stream>>>(a3s, a3d, ei, E, Etot, e3, m3);
  l3_exp_sum<<<blk(Etot, B), B, 0, stream>>>(m3, ei, E, Etot, e3, s3);
  l3_aggregate<<<blk(Etot, B), B, 0, stream>>>(h3, e3, s3, ei, E, Etot, agg3);
  final_log_softmax<<<blk(N, B), B, 0, stream>>>(agg3, b3, (float*)d_out, N);

  (void)n_in; (void)out_size; (void)ws_size;
}
